// SAGEEncoder_55499567399320
// MI455X (gfx1250) — matrix operand order repaired and checked by reading, not compiled, not run on hardware
//
#include <hip/hip_runtime.h>
#include <stdint.h>

#define N_NODES 50000
#define N_EDGES 640000

typedef float v8f __attribute__((ext_vector_type(8)));
typedef __bf16 v16bf __attribute__((ext_vector_type(16)));
typedef unsigned short v16u __attribute__((ext_vector_type(16)));
typedef unsigned short v8u  __attribute__((ext_vector_type(8)));

// fp32 -> bf16 round-to-nearest-even (bit pattern)
static __device__ __forceinline__ unsigned short f2bf(float f) {
  unsigned u = __builtin_bit_cast(unsigned, f);
  u += 0x7fffu + ((u >> 16) & 1u);
  return (unsigned short)(u >> 16);
}

static __device__ __forceinline__ void fatomic_add(float* p, float v) {
  __hip_atomic_fetch_add(p, v, __ATOMIC_RELAXED, __HIP_MEMORY_SCOPE_AGENT);
}

// ---------------------------------------------------------------- zero
__global__ void k_zero(float4* __restrict__ p, int n4) {
  int i = blockIdx.x * blockDim.x + threadIdx.x;
  if (i < n4) { float4 z = {0.f, 0.f, 0.f, 0.f}; p[i] = z; }
}

// ------------------------------------------- pack weights -> bf16 B-fragments
// Layout per (nt:16x16-col-tile, kt:32-K-tile): [nt(16)][kt(8)][lane(32)][elem(16)]
// B fragment (16-bit, 32x16): lane L -> column N = n0 + L%16, khi = L/16,
//                             element j -> K = k0 + khi*16 + j      (this code's order; A uses the same)
// B1[k][n] = k<128 ? w1_l[k][n] : w1_r[k-128][n]           (256x256)
// B2[k][n] = n<128 ? w2_l[k][n] : w2_r[k][n-128]           (256x256)
__global__ void k_pack(const float* __restrict__ w1l, const float* __restrict__ w1r,
                       const float* __restrict__ w2l, const float* __restrict__ w2r,
                       unsigned short* __restrict__ b1p, unsigned short* __restrict__ b2p) {
  int idx = blockIdx.x * blockDim.x + threadIdx.x;   // 0 .. 2*65536-1
  int which = idx >> 16;
  int t = idx & 65535;
  int j    =  t        & 15;
  int lane = (t >>  4) & 31;
  int kt   = (t >>  9) &  7;
  int nt   =  t >> 12;
  int n   = nt * 16 + (lane & 15);
  int khi = lane >> 4;
  int k   = kt * 32 + khi * 16 + j;
  float v;
  if (which == 0) v = (k < 128) ? w1l[k * 256 + n] : w1r[(k - 128) * 256 + n];
  else            v = (n < 128) ? w2l[k * 128 + n] : w2r[k * 128 + (n - 128)];
  (which ? b2p : b1p)[t] = f2bf(v);
}

// ---------------------------------------------------------------- degrees
__global__ void k_degree(const long long* __restrict__ dst, float* __restrict__ cnt) {
  int e = blockIdx.x * blockDim.x + threadIdx.x;
  if (e < N_EDGES) fatomic_add(&cnt[dst[e]], 1.0f);
}

// -------------------------------- scatter-add 128 channels (1 edge per wave)
__global__ void k_scatter(const long long* __restrict__ src, const long long* __restrict__ dst,
                          const float* __restrict__ feat, int stride, float* __restrict__ agg) {
  int tid = blockIdx.x * blockDim.x + threadIdx.x;
  int e = tid >> 5;
  if (e >= N_EDGES) return;
  int c = (tid & 31) * 4;
  long long s = src[e], d = dst[e];
  float4 v = *(const float4*)(feat + (size_t)s * stride + c);
  float* b = agg + (size_t)d * 128 + c;
  fatomic_add(b + 0, v.x);
  fatomic_add(b + 1, v.y);
  fatomic_add(b + 2, v.z);
  fatomic_add(b + 3, v.w);
}

// -------- prep: abf[n] = bf16([agg1[n]/cnt[n] | x[n]])  (N x 256 bf16 A-matrix)
__global__ void k_prep(const float* __restrict__ agg1, const float* __restrict__ cnt,
                       const float* __restrict__ x, unsigned short* __restrict__ abf) {
  int tid = blockIdx.x * blockDim.x + threadIdx.x;
  if (tid >= N_NODES * 128) return;
  int n = tid >> 7, c = tid & 127;
  float recip = 1.0f / fmaxf(cnt[n], 1.0f);
  abf[(size_t)n * 256 + c]       = f2bf(agg1[tid] * recip);
  abf[(size_t)n * 256 + 128 + c] = f2bf(x[tid]);
}

// ---------------- unified bf16 WMMA GEMM: D = A(Nx256) @ B(256x256)
// Per-wave 16x16 output tile, K=256 in 8 x wmma_f32_16x16x32_bf16 steps.
// MODE 0: D = bf16(relu(acc + bias))   (layer-1, writes hbf)
// MODE 1: D = fp32 acc                 (layer-2, writes q)
template <int MODE>
__global__ void __launch_bounds__(256)
k_gemm(const unsigned short* __restrict__ A,   // N x 256 bf16, row-major
       const unsigned short* __restrict__ Bp,  // packed B fragments
       const float* __restrict__ bias,         // 256 (MODE 0 only)
       unsigned short* __restrict__ outBf,     // MODE 0
       float* __restrict__ outF) {             // MODE 1
  int wave = threadIdx.x >> 5;
  int lane = threadIdx.x & 31;
  int t  = blockIdx.x * 8 + wave;          // 50000 tiles (3125 x 16)
  int mt = t >> 4, nt = t & 15;
  int m0 = mt * 16;
  int khi  = lane >> 4;
  int mrow = m0 + (lane & 15);
  const unsigned short* aRow = A + (size_t)mrow * 256;
  const unsigned short* bTile = Bp + ((size_t)(nt << 3) << 9) + (lane << 4);

  v8f acc = {};
#pragma unroll
  for (int kt = 0; kt < 8; ++kt) {
    // A fragment: elem j -> K = kt*32 + khi*16 + j  (two 8-elem runs)
    int base0 = kt * 32 + khi * 16;
    v8u r0 = *(const v8u*)(aRow + base0);        // K = base0 .. +7
    v8u r1 = *(const v8u*)(aRow + base0 + 8);    // K = base0+8 .. +15
    v16u au = __builtin_shufflevector(r0, r1, 0, 1, 2, 3, 4, 5, 6, 7,
                                              8, 9, 10, 11, 12, 13, 14, 15);
    v16u bu = *(const v16u*)(bTile + (kt << 9));
    acc = __builtin_amdgcn_wmma_f32_16x16x32_bf16(
        false, __builtin_bit_cast(v16bf, au),
        false, __builtin_bit_cast(v16bf, bu),
        (short)0, acc, false, false);
  }

  int col = nt * 16 + (lane & 15);
  if (MODE == 0) {
    float bv = bias[col];
#pragma unroll
    for (int r = 0; r < 8; ++r) {            // C/D: VGPR r, lane -> M = m0 + r + 8*khi
      int M = m0 + r + 8 * khi;
      outBf[(size_t)M * 256 + col] = f2bf(fmaxf(acc[r] + bv, 0.0f));
    }
  } else {
#pragma unroll
    for (int r = 0; r < 8; ++r) {
      int M = m0 + r + 8 * khi;
      outF[(size_t)M * 256 + col] = acc[r];
    }
  }
}

// ------------- final: out = agg2/cnt + b2_l + q[:,128:]
__global__ void k_final(const float* __restrict__ agg2, const float* __restrict__ cnt,
                        const float* __restrict__ b2l, const float* __restrict__ q,
                        float* __restrict__ out) {
  int tid = blockIdx.x * blockDim.x + threadIdx.x;
  if (tid >= N_NODES * 128) return;
  int n = tid >> 7, c = tid & 127;
  float recip = 1.0f / fmaxf(cnt[n], 1.0f);
  out[tid] = agg2[tid] * recip + b2l[c] + q[(size_t)n * 256 + 128 + c];
}

// ---------------------------------------------------------------- launch
extern "C" void kernel_launch(void* const* d_in, const int* in_sizes, int n_in,
                              void* d_out, int out_size, void* d_ws, size_t ws_size,
                              hipStream_t stream) {
  const float*     x   = (const float*)d_in[0];
  const long long* ei  = (const long long*)d_in[1];   // int64 edge_index [2, E]
  const float*     w1l = (const float*)d_in[2];
  const float*     b1l = (const float*)d_in[3];
  const float*     w1r = (const float*)d_in[4];
  const float*     w2l = (const float*)d_in[5];
  const float*     b2l = (const float*)d_in[6];
  const float*     w2r = (const float*)d_in[7];
  float* out = (float*)d_out;

  char* ws = (char*)d_ws;
  // workspace layout (hbf aliases agg1: agg1 is dead once abf is built)
  float*          cnt  = (float*)(ws + 0);                    //   200,192 B
  float*          agg1 = (float*)(ws + 200192);               // 25.6 MB
  unsigned short* hbf  = (unsigned short*)(ws + 200192);      // 25.6 MB (alias agg1)
  float*          agg2 = (float*)(ws + 25800192);             // 25.6 MB
  unsigned short* abf  = (unsigned short*)(ws + 51400192);    // 25.6 MB (bf16 [mean|x])
  float*          q    = (float*)(ws + 77000192);             // 51.2 MB
  unsigned short* b1p  = (unsigned short*)(ws + 128200192);   // 128 KB
  unsigned short* b2p  = (unsigned short*)(ws + 128331264);   // 128 KB

  const long long* src = ei;
  const long long* dst = ei + N_EDGES;

  // 1. zero cnt + agg1 + agg2 (contiguous 51,400,192 bytes)
  int n4 = 51400192 / 16;
  k_zero<<<(n4 + 255) / 256, 256, 0, stream>>>((float4*)ws, n4);

  // 2. pack both weight stacks into bf16 WMMA-B fragment order
  k_pack<<<512, 256, 0, stream>>>(w1l, w1r, w2l, w2r, b1p, b2p);

  // 3. in-degree counts
  k_degree<<<(N_EDGES + 255) / 256, 256, 0, stream>>>(dst, cnt);

  // 4. layer-1 scatter: agg1[dst] += x[src]   (128 ch)
  k_scatter<<<(N_EDGES * 32) / 256, 256, 0, stream>>>(src, dst, x, 128, agg1);

  // 5. abf = bf16([agg1/cnt | x])  -- hoists all cvt/scale out of the GEMM loop
  k_prep<<<(N_NODES * 128 + 255) / 256, 256, 0, stream>>>(agg1, cnt, x, abf);

  // 6. h = relu(abf @ B1 + b1)                (pure-WMMA loop, bf16 out)
  k_gemm<0><<<6250, 256, 0, stream>>>(abf, b1p, b1l, hbf, nullptr);

  // 7. q = h @ [w2_l | w2_r]                  (transform BEFORE aggregate)
  k_gemm<1><<<6250, 256, 0, stream>>>(hbf, b2p, nullptr, nullptr, q);

  // 8. layer-2 scatter on the already-transformed 128-ch half
  k_scatter<<<(N_EDGES * 32) / 256, 256, 0, stream>>>(src, dst, q, 256, agg2);

  // 9. out = agg2/cnt + b2 + q[:,128:]
  k_final<<<(N_NODES * 128 + 255) / 256, 256, 0, stream>>>(agg2, cnt, b2l, q, out);
}
